// sLSTMCell_86500641341529
// MI455X (gfx1250) — matrix operand order repaired and checked by reading, not compiled, not run on hardware
//
#include <hip/hip_runtime.h>
#include <stdint.h>

// sLSTM cell for MI455X (gfx1250, wave32, WMMA).
// B=131072, IN=H=128. gates = [x|h] @ [W_x ; W_h + r] + b  (r folded into W once).
// bf16 WMMA (16x16x32) with f32 accumulation keeps the GEMM (~34 GFLOP) far under
// the 576 MB / 23.3 TB/s ~ 25us HBM floor, so the kernel is memory-bound. All five
// input streams (x,h,c,n,m) are issued before the GEMM so their HBM latency hides
// under compute; outputs are streamed with non-temporal hints.

typedef __attribute__((ext_vector_type(16))) __bf16 v16bf;
typedef __attribute__((ext_vector_type(8)))  float  v8f;

static __device__ __forceinline__ uint16_t f32_to_bf16_rne(float f) {
  union { float f; uint32_t u; } v; v.f = f;
  uint32_t u = v.u;
  u += 0x7FFFu + ((u >> 16) & 1u);   // round-to-nearest-even
  return (uint16_t)(u >> 16);
}

// ---------------------------------------------------------------------------
// Prep: build combined weights Bc[256,512] = [W[0:128,:] ; W[128:256,:] + r],
// converted to bf16 and pre-swizzled into a WMMA B-fragment layout using this kernel's k order (same as A):
//   tile (kt in 0..7, nt in 0..31), 32x16 bf16 per tile, 512 ushorts per tile:
//   element (kk,nn):  lane = nn + 16*(kk>=16), slot = kk&15
//   -> per-lane 32 contiguous bytes, loadable as 2x global_load_b128.
// 256 KB total, L2-resident (192 MB L2) for the whole main kernel.
// ---------------------------------------------------------------------------
__global__ __launch_bounds__(256) void slstm_prep_weights(
    const float* __restrict__ W, const float* __restrict__ r,
    uint16_t* __restrict__ Bc) {
  int e = blockIdx.x * 256 + threadIdx.x;   // 0 .. 131071
  int k = e >> 9;                            // 0..255
  int n = e & 511;                           // 0..511
  float v = W[k * 512 + n];
  if (k >= 128) v += r[(k - 128) * 512 + n];
  int kt = k >> 5, kk = k & 31;
  int nt = n >> 4, nn = n & 15;
  int lane = nn + ((kk >> 4) << 4);
  int slot = kk & 15;
  Bc[(kt * 32 + nt) * 512 + lane * 16 + slot] = f32_to_bf16_rne(v);
}

// ---------------------------------------------------------------------------
// Main: one block = 16 batch rows x all 512 gate columns. 8 waves, each wave
// owns a 16x64 strip (4 accumulator tiles), K-loop 8 x 32 over [x|h].
// ---------------------------------------------------------------------------
#define ROWS 16

__global__ __launch_bounds__(256) void slstm_main(
    const float* __restrict__ x, const float* __restrict__ h,
    const float* __restrict__ c, const float* __restrict__ n,
    const float* __restrict__ m, const float* __restrict__ b,
    const uint16_t* __restrict__ Bc, float* __restrict__ out, int Btot) {
  // A tiles in WMMA A-fragment layout: 8 K-steps x (32 lanes x 16 bf16) = 8 KB
  __shared__ __align__(32) uint16_t Als[8 * 512];
  // Gate accumulators, padded row stride to break LDS bank conflicts (33 KB)
  __shared__ float gl[ROWS][512 + 4];

  const int tid = threadIdx.x;
  const int rowbase = blockIdx.x * ROWS;

  // Elementwise geometry: e = tid + s*256, and 256 % 128 == 0, so each thread
  // owns ONE gate column j for all its 8 elements; rows step by 2.
  const int jcol = tid & 127;
  const int row0 = tid >> 7;                 // 0 or 1

  // ---- (1) Issue the x|h staging loads (oldest in the LOADcnt queue) ----
  float av[16];
#pragma unroll
  for (int s = 0; s < 16; ++s) {
    int e = tid + s * 256;            // 0..4095
    int row = e >> 8;                 // 0..15
    int k = e & 255;                  // 0..255 (coalesced along k)
    av[s] = (k < 128) ? x[(rowbase + row) * 128 + k]
                      : h[(rowbase + row) * 128 + (k - 128)];
  }

  // ---- (2) Issue c/n/m (and bias) loads now: issued after the staging loads,
  // so staging's waits don't drain them; they complete under the GEMM. ----
  float cv[8], nv[8], mv[8];
#pragma unroll
  for (int s = 0; s < 8; ++s) {
    int row = row0 + 2 * s;
    long long g = (long long)(rowbase + row) * 128 + jcol;
    cv[s] = __builtin_nontemporal_load(&c[g]);
    nv[s] = __builtin_nontemporal_load(&n[g]);
    mv[s] = __builtin_nontemporal_load(&m[g]);
  }
  const float bi = b[jcol];
  const float bf = b[jcol + 128];
  const float bz = b[jcol + 256];
  const float bo = b[jcol + 384];

  // ---- (3) Convert fp32 -> bf16 and store into LDS in A-fragment layout ----
  // A-frag (16x32 bf16), this kernel's k order (same as Bc): lane = row + 16*(kk>>4), slot = kk&15
#pragma unroll
  for (int s = 0; s < 16; ++s) {
    int e = tid + s * 256;
    int row = e >> 8;
    int k = e & 255;
    int kt = k >> 5, kk = k & 31;
    int lane = row + ((kk >> 4) << 4);
    int slot = kk & 15;
    Als[kt * 512 + lane * 16 + slot] = f32_to_bf16_rne(av[s]);
  }
  __syncthreads();

  const int wave = tid >> 5;
  const int lane = tid & 31;
  const int ntbase = wave * 4;        // 4 n-tiles (64 columns) per wave

  v8f acc[4] = {};

  // ---- GEMM: 8 K-steps x 4 N-tiles of v_wmma_f32_16x16x32_bf16 ----
#pragma unroll
  for (int kt = 0; kt < 8; ++kt) {
    const v16bf a = *(const v16bf*)(&Als[kt * 512 + lane * 16]);
#pragma unroll
    for (int j = 0; j < 4; ++j) {
      const v16bf bt =
          *(const v16bf*)(&Bc[(kt * 32 + ntbase + j) * 512 + lane * 16]);
      acc[j] = __builtin_amdgcn_wmma_f32_16x16x32_bf16(
          false, a, false, bt, (short)0, acc[j], false, false);
    }
  }

  // ---- Epilogue: C layout (lane<16: M=v, N=lane; lane>=16: M=8+v, N=lane-16)
  {
    const int col0 = wave * 64 + (lane & 15);
    const int mrow = (lane >> 4) * 8;
#pragma unroll
    for (int j = 0; j < 4; ++j) {
#pragma unroll
      for (int v = 0; v < 8; ++v) {
        gl[mrow + v][col0 + j * 16] = acc[j][v];
      }
    }
  }
  __syncthreads();

  // ---- Elementwise sLSTM update (fp32, faithful to reference) ----
  const long long BH = (long long)Btot * 128;
#pragma unroll
  for (int s = 0; s < 8; ++s) {
    int row = row0 + 2 * s;
    long long g = (long long)(rowbase + row) * 128 + jcol;

    float i_raw = gl[row][jcol]       + bi;
    float f_raw = gl[row][jcol + 128] + bf;
    float z_raw = gl[row][jcol + 256] + bz;
    float o_raw = gl[row][jcol + 384] + bo;

    float f_g    = 1.0f / (1.0f + __expf(-f_raw));      // sigmoid
    float z_g    = tanhf(z_raw);
    float o_g    = 1.0f / (1.0f + __expf(-o_raw));
    float log_f  = -log1pf(__expf(-f_raw));             // log(sigmoid(f_raw))

    float m_new = fmaxf(log_f + mv[s], i_raw);          // log(i) == i_raw
    float i_p   = __expf(i_raw - m_new);
    // f' = exp(log f + m_new - m_new) == f  (as in reference source)
    float c_new = f_g * cv[s] + i_p * z_g;
    float n_new = f_g * nv[s] + i_p;
    float h_new = o_g * tanhf(c_new / n_new);

    __builtin_nontemporal_store(h_new, &out[g]);
    __builtin_nontemporal_store(c_new, &out[BH + g]);
    __builtin_nontemporal_store(n_new, &out[2 * BH + g]);
    __builtin_nontemporal_store(m_new, &out[3 * BH + g]);
  }
}

extern "C" void kernel_launch(void* const* d_in, const int* in_sizes, int n_in,
                              void* d_out, int out_size, void* d_ws, size_t ws_size,
                              hipStream_t stream) {
  const float* x = (const float*)d_in[0];
  const float* h = (const float*)d_in[1];
  const float* c = (const float*)d_in[2];
  const float* n = (const float*)d_in[3];
  const float* m = (const float*)d_in[4];
  const float* W = (const float*)d_in[5];
  const float* r = (const float*)d_in[6];
  const float* b = (const float*)d_in[7];
  float* out = (float*)d_out;
  uint16_t* Bc = (uint16_t*)d_ws;           // 256 KB swizzled bf16 weights

  const int Btot = in_sizes[0] / 128;       // 131072

  // Fold r into W, convert to bf16, pre-swizzle into WMMA B-fragment layout.
  slstm_prep_weights<<<(256 * 512) / 256, 256, 0, stream>>>(W, r, Bc);

  // Fused GEMM + sLSTM elementwise update.
  slstm_main<<<Btot / ROWS, 256, 0, stream>>>(x, h, c, n, m, b, Bc, out, Btot);
}
